// ImageEncoder_21191368639300
// MI455X (gfx1250) — compile-verified, matrix operand order checked by reading, not run on hardware
//
#include <hip/hip_runtime.h>
#include <math.h>

// ---------------------------------------------------------------------------
// Transformer encoder (N=4,S=2048,D=512,H=8,HD=64,L=4,F=200) for gfx1250.
// Matrix-bound (~220 GFLOP, data fits in 192MB L2) => bf16 WMMA with f32
// accumulate. All fragments are contiguous per-lane LDS chunks (ds_load_b128).
// Attention: softmax row-sums computed BY the matrix engine (ones-column in
// V tile); max reduction uses DPP + 2 bpermutes only.
// ---------------------------------------------------------------------------

typedef __bf16 bf16_t;
typedef __bf16 v16bf __attribute__((ext_vector_type(16)));
typedef __bf16 v8bf  __attribute__((ext_vector_type(8)));
typedef float  v8f   __attribute__((ext_vector_type(8)));

__device__ __forceinline__ v8f wmma_bf16(v16bf a, v16bf b, v8f c) {
  return __builtin_amdgcn_wmma_f32_16x16x32_bf16(false, a, false, b,
                                                 (short)0, c, false, false);
}

// Fragment from row-major LDS tile [16 rows][32 k] (stride ld elems).
// Per-lane data = two contiguous 16B chunks -> ds_load_b128 pairs.
__device__ __forceinline__ v16bf frag_rm(const bf16_t* base, int ld, int lane) {
  const int half = lane >> 4, m = lane & 15;
  const bf16_t* p = base + m * ld + half * 8;
  v16bf f;
#pragma unroll
  for (int e = 0; e < 8; ++e) f[e] = p[e];
#pragma unroll
  for (int e = 0; e < 8; ++e) f[8 + e] = p[16 + e];
  return f;
}

// max with lane^1 / lane^2 neighbor via DPP quad_perm (VALU, no DS traffic).
// CTRL is a template arg: __builtin_amdgcn_update_dpp needs a constant.
template <int CTRL>
__device__ __forceinline__ float max_dpp(float x) {
  int xi = __builtin_bit_cast(int, x);
  int yi = __builtin_amdgcn_update_dpp(xi, xi, CTRL, 0xF, 0xF, true);
  return fmaxf(x, __builtin_bit_cast(float, yi));
}

// ---------------------------------------------------------------------------
// Elementwise / setup kernels
// ---------------------------------------------------------------------------
__global__ __launch_bounds__(256) void pos_encode(const float* __restrict__ feat,
                                                  float* __restrict__ xF,
                                                  bf16_t* __restrict__ xB,
                                                  int total, int S_, int Dd) {
  int i = blockIdx.x * 256 + threadIdx.x;
  if (i >= total) return;
  int d = i % Dd;
  int s = (i / Dd) % S_;
  int pair = d >> 1;
  float freq = __expf(-(float)(2 * pair) * (logf(10000.f) / (float)Dd));
  float ang = (float)s * freq;
  float pe = (d & 1) ? __cosf(ang) : __sinf(ang);
  float v = feat[i] + pe;
  xF[i] = v;
  xB[i] = (bf16_t)v;
}

// f32 [K][N] -> bf16 [N][K] (per-layer via blockIdx.z), LDS-tiled transpose
__global__ __launch_bounds__(256) void cvt_transpose(const float* __restrict__ src,
                                                     bf16_t* __restrict__ dst,
                                                     int K, int N) {
  __shared__ float tile[32][33];
  const float* s = src + (size_t)blockIdx.z * K * N;
  bf16_t* d = dst + (size_t)blockIdx.z * K * N;
  const int n0 = blockIdx.x * 32, k0 = blockIdx.y * 32;
  const int tx = threadIdx.x & 31, ty = threadIdx.x >> 5;
#pragma unroll
  for (int i = 0; i < 32; i += 8) {
    int k = k0 + ty + i, n = n0 + tx;
    tile[ty + i][tx] = (k < K && n < N) ? s[(size_t)k * N + n] : 0.f;
  }
  __syncthreads();
#pragma unroll
  for (int i = 0; i < 32; i += 8) {
    int n = n0 + ty + i, k = k0 + tx;
    if (n < N && k < K) d[(size_t)n * K + k] = (bf16_t)tile[tx][ty + i];
  }
}

// wave-per-row layernorm: f32 in -> f32 + bf16 out
__global__ __launch_bounds__(256) void layernorm(const float* __restrict__ in,
                                                 const float* __restrict__ w,
                                                 const float* __restrict__ b,
                                                 float* __restrict__ outF,
                                                 bf16_t* __restrict__ outB,
                                                 int rows, int Dd) {
  int row = blockIdx.x * 8 + (threadIdx.x >> 5);
  int lane = threadIdx.x & 31;
  if (row >= rows) return;
  const float* x = in + (size_t)row * Dd;
  float s = 0.f, ss = 0.f;
  for (int i = lane; i < Dd; i += 32) { float v = x[i]; s += v; ss += v * v; }
#pragma unroll
  for (int off = 16; off > 0; off >>= 1) {
    s  += __shfl_xor(s,  off, 32);
    ss += __shfl_xor(ss, off, 32);
  }
  float mean = s / (float)Dd;
  float var = ss / (float)Dd - mean * mean;
  float inv = rsqrtf(var + 1e-5f);
  for (int i = lane; i < Dd; i += 32) {
    float v = (x[i] - mean) * inv * w[i] + b[i];
    outF[(size_t)row * Dd + i] = v;
    outB[(size_t)row * Dd + i] = (bf16_t)v;
  }
}

// ---------------------------------------------------------------------------
// bf16 WMMA GEMM: C[M,N] = A[M,K] @ B[K,N], B given TRANSPOSED as Bt[N][K].
// Block 128 thr = 4 waves; block tile 64x128; wave tile 32x64 (8 WMMA/step).
// ALIGNED=true: M%64==0, Nn%128==0, Kk%32==0 -> no guards anywhere.
// ---------------------------------------------------------------------------
#define GM_BM 64
#define GM_BN 128
#define GM_BK 32
#define TSTR  40

template <bool ALIGNED>
__global__ __launch_bounds__(128) void gemm_bf16(
    const bf16_t* __restrict__ A, const bf16_t* __restrict__ Bt,
    const float* __restrict__ bias, const float* __restrict__ resid,
    float* __restrict__ outF, bf16_t* __restrict__ outB,
    bf16_t* __restrict__ outBT,
    int M, int Nn, int Kk, int relu) {
  __shared__ __align__(16) bf16_t Asm[GM_BM * TSTR];
  __shared__ __align__(16) bf16_t Bsm[GM_BN * TSTR];

  const int tid = threadIdx.x;
  const int wave = tid >> 5, lane = tid & 31;
  const int m0 = blockIdx.y * GM_BM, n0 = blockIdx.x * GM_BN;
  const int wm = (wave >> 1) * 32, wn = (wave & 1) * 64;

  v8f acc[2][4] = {};

  for (int k0 = 0; k0 < Kk; k0 += GM_BK) {
    __syncthreads();
    for (int idx = tid; idx < (GM_BM * GM_BK) / 8; idx += 128) {
      int row = idx >> 2, col = (idx & 3) * 8;
      int gr = m0 + row, gc = k0 + col;
      bf16_t* dstp = &Asm[row * TSTR + col];
      if (ALIGNED) {
        *(uint4*)dstp = *(const uint4*)(A + (size_t)gr * Kk + gc);
      } else if (gr < M && gc + 8 <= Kk) {
        *(uint4*)dstp = *(const uint4*)(A + (size_t)gr * Kk + gc);
      } else {
#pragma unroll
        for (int j = 0; j < 8; ++j)
          dstp[j] = (gr < M && gc + j < Kk) ? A[(size_t)gr * Kk + gc + j] : (bf16_t)0.f;
      }
    }
    for (int idx = tid; idx < (GM_BN * GM_BK) / 8; idx += 128) {
      int row = idx >> 2, col = (idx & 3) * 8;
      int gr = n0 + row, gc = k0 + col;
      bf16_t* dstp = &Bsm[row * TSTR + col];
      if (ALIGNED) {
        *(uint4*)dstp = *(const uint4*)(Bt + (size_t)gr * Kk + gc);
      } else if (gr < Nn && gc + 8 <= Kk) {
        *(uint4*)dstp = *(const uint4*)(Bt + (size_t)gr * Kk + gc);
      } else {
#pragma unroll
        for (int j = 0; j < 8; ++j)
          dstp[j] = (gr < Nn && gc + j < Kk) ? Bt[(size_t)gr * Kk + gc + j] : (bf16_t)0.f;
      }
    }
    __syncthreads();

    if (k0 + GM_BK < Kk) {
      const int pr = tid >> 1, pc = k0 + GM_BK + (tid & 1) * 8;
      __builtin_prefetch(A  + (size_t)(m0 + (pr & (GM_BM - 1))) * Kk + pc, 0, 0);
      __builtin_prefetch(Bt + (size_t)(n0 + pr) * Kk + pc, 0, 0);
    }

    v16bf aF[2], bF[4];
    aF[0] = frag_rm(&Asm[(wm + 0)  * TSTR], TSTR, lane);
    aF[1] = frag_rm(&Asm[(wm + 16) * TSTR], TSTR, lane);
#pragma unroll
    for (int ni = 0; ni < 4; ++ni)
      bF[ni] = frag_rm(&Bsm[(wn + ni * 16) * TSTR], TSTR, lane);
#pragma unroll
    for (int mi = 0; mi < 2; ++mi)
#pragma unroll
      for (int ni = 0; ni < 4; ++ni)
        acc[mi][ni] = wmma_bf16(aF[mi], bF[ni], acc[mi][ni]);
  }

  const int half = lane >> 4, nl = lane & 15;
#pragma unroll
  for (int mi = 0; mi < 2; ++mi)
#pragma unroll
    for (int ni = 0; ni < 4; ++ni) {
      const int gmb = m0 + wm + mi * 16 + half * 8;
      const int gn  = n0 + wn + ni * 16 + nl;
      if (!ALIGNED && gn >= Nn) continue;
      float vv[8];
#pragma unroll
      for (int r = 0; r < 8; ++r) {
        float v = acc[mi][ni][r];
        if (bias) v += bias[gn];
        if (relu) v = fmaxf(v, 0.f);
        vv[r] = v;
      }
      if (outBT) {  // [N][M]: lane's 8 rows are contiguous -> one 16B store
        v8bf pk;
#pragma unroll
        for (int r = 0; r < 8; ++r) pk[r] = (bf16_t)vv[r];
        *(uint4*)(outBT + (size_t)gn * M + gmb) = __builtin_bit_cast(uint4, pk);
      }
#pragma unroll
      for (int r = 0; r < 8; ++r) {
        int gm = gmb + r;
        if (ALIGNED || gm < M) {
          float v = vv[r];
          if (resid) v += resid[(size_t)gm * Nn + gn];
          if (outF) outF[(size_t)gm * Nn + gn] = v;
          if (outB) outB[(size_t)gm * Nn + gn] = (bf16_t)v;
        }
      }
    }
}

// ---------------------------------------------------------------------------
// Flash attention. V PRE-TRANSPOSED: Vt[D][N*S]. Key block = 64.
// Block = (64 queries, head, batch); 4 waves, wave = 16 query rows.
// Softmax row-sums come from the matrix engine via a ones-row block in the
// V tile (accumulator tile o[4]); only the row-max needs cross-lane ops.
// ---------------------------------------------------------------------------
#define AT_STR 72

__global__ __launch_bounds__(128) void attention(
    const bf16_t* __restrict__ Q, const bf16_t* __restrict__ K,
    const bf16_t* __restrict__ Vt, bf16_t* __restrict__ Y,
    int S_, int Dtot, int NS_) {
  const int qb   = blockIdx.x * 64;
  const int h    = blockIdx.y;
  const int bat  = blockIdx.z;
  const int tid  = threadIdx.x;
  const int wave = tid >> 5, lane = tid & 31;
  const int half = lane >> 4, nl = lane & 15;
  const size_t rowbase = (size_t)bat * S_;
  const int colbase = h * 64;

  __shared__ __align__(16) bf16_t Qs[64 * AT_STR];
  __shared__ __align__(16) bf16_t Ks[64 * AT_STR];       // [key][dim]
  __shared__ __align__(16) bf16_t Vts[80 * AT_STR];      // [dim][key]; rows 64..79 = ones block
  __shared__ __align__(16) bf16_t Ps[4 * 16 * AT_STR];   // per-wave P tiles [16][64]

  // Q tile 64x64 + ones-block init (rows 64..79 of Vts: row 64 = 1, rest 0)
  for (int idx = tid; idx < (64 * 64) / 8; idx += 128) {
    int row = idx >> 3, col = (idx & 7) * 8;
    *(uint4*)&Qs[row * AT_STR + col] =
        *(const uint4*)(Q + (rowbase + qb + row) * Dtot + colbase + col);
  }
  for (int idx = tid; idx < 16 * AT_STR; idx += 128) {
    int row = idx / AT_STR, col = idx % AT_STR;
    Vts[(64 + row) * AT_STR + col] = (row == 0) ? (bf16_t)1.f : (bf16_t)0.f;
  }
  __syncthreads();

  v16bf aQ0 = frag_rm(&Qs[wave * 16 * AT_STR + 0],  AT_STR, lane);
  v16bf aQ1 = frag_rm(&Qs[wave * 16 * AT_STR + 32], AT_STR, lane);
  v16bf bOne0 = frag_rm(&Vts[64 * AT_STR + 0],  AT_STR, lane);
  v16bf bOne1 = frag_rm(&Vts[64 * AT_STR + 32], AT_STR, lane);

  v8f o[5] = {};          // o[0..3] = output dim tiles, o[4] = softmax row sums
  float mstat[8];
#pragma unroll
  for (int r = 0; r < 8; ++r) mstat[r] = -3.0e38f;

  bf16_t* PsW = &Ps[wave * 16 * AT_STR];
  const float scale = 0.125f;  // 1/sqrt(64)

  for (int kb = 0; kb < S_; kb += 64) {
    __syncthreads();
    for (int idx = tid; idx < (64 * 64) / 8; idx += 128) {
      int row = idx >> 3, col = (idx & 7) * 8;
      *(uint4*)&Ks[row * AT_STR + col] =
          *(const uint4*)(K + (rowbase + kb + row) * Dtot + colbase + col);
      *(uint4*)&Vts[row * AT_STR + col] =
          *(const uint4*)(Vt + (size_t)(colbase + row) * NS_ + rowbase + kb + col);
    }
    __syncthreads();

    // scores: Q @ K^T, 4 key subtiles
    v8f sc[4];
#pragma unroll
    for (int nt = 0; nt < 4; ++nt) {
      v16bf b0 = frag_rm(&Ks[nt * 16 * AT_STR + 0],  AT_STR, lane);
      v16bf b1 = frag_rm(&Ks[nt * 16 * AT_STR + 32], AT_STR, lane);
      v8f c = {};
      c = wmma_bf16(aQ0, b0, c);
      c = wmma_bf16(aQ1, b1, c);
      sc[nt] = c * scale;
    }

    // online rescale: only the MAX needs cross-lane reduction
#pragma unroll
    for (int r = 0; r < 8; ++r) {
      float bm = fmaxf(fmaxf(sc[0][r], sc[1][r]), fmaxf(sc[2][r], sc[3][r]));
      bm = max_dpp<0xB1>(bm);             // lane ^ 1 (quad_perm [1,0,3,2])
      bm = max_dpp<0x4E>(bm);             // lane ^ 2 (quad_perm [2,3,0,1])
      bm = fmaxf(bm, __shfl_xor(bm, 4, 32));
      bm = fmaxf(bm, __shfl_xor(bm, 8, 32));
      float mnew  = fmaxf(mstat[r], bm);
      float alpha = __expf(mstat[r] - mnew);
      mstat[r] = mnew;
#pragma unroll
      for (int nt = 0; nt < 4; ++nt) sc[nt][r] = __expf(sc[nt][r] - mnew);
#pragma unroll
      for (int t = 0; t < 5; ++t) o[t][r] *= alpha;
    }

    // P (C-layout) -> per-wave LDS tile [16][64], re-pack as A fragments
#pragma unroll
    for (int nt = 0; nt < 4; ++nt)
#pragma unroll
      for (int r = 0; r < 8; ++r)
        PsW[(r + half * 8) * AT_STR + nt * 16 + nl] = (bf16_t)sc[nt][r];
    asm volatile("s_wait_dscnt 0" ::: "memory");

    v16bf aP0 = frag_rm(PsW + 0,  AT_STR, lane);
    v16bf aP1 = frag_rm(PsW + 32, AT_STR, lane);
#pragma unroll
    for (int nt = 0; nt < 4; ++nt) {
      o[nt] = wmma_bf16(aP0, frag_rm(&Vts[nt * 16 * AT_STR + 0],  AT_STR, lane), o[nt]);
      o[nt] = wmma_bf16(aP1, frag_rm(&Vts[nt * 16 * AT_STR + 32], AT_STR, lane), o[nt]);
    }
    o[4] = wmma_bf16(aP0, bOne0, o[4]);   // row-sum accumulation (ones block)
    o[4] = wmma_bf16(aP1, bOne1, o[4]);
  }

  // normalize: row sums live in column 0 of o[4] (lane half*16)
#pragma unroll
  for (int r = 0; r < 8; ++r) {
    float ls = __shfl(o[4][r], half * 16, 32);
    float inv = 1.0f / ls;
#pragma unroll
    for (int nt = 0; nt < 4; ++nt) {
      int gm = qb + wave * 16 + r + half * 8;
      Y[(rowbase + gm) * Dtot + colbase + nt * 16 + nl] = (bf16_t)(o[nt][r] * inv);
    }
  }
}

// ---------------------------------------------------------------------------
// Host-side pipeline
// ---------------------------------------------------------------------------
extern "C" void kernel_launch(void* const* d_in, const int* in_sizes, int n_in,
                              void* d_out, int out_size, void* d_ws, size_t ws_size,
                              hipStream_t stream) {
  (void)in_sizes; (void)n_in; (void)out_size; (void)ws_size;
  constexpr int NB = 4, S = 2048, D = 512, H = 8, LAY = 4, F = 200;
  constexpr int NS = NB * S;

  const float* feat = (const float*)d_in[0];
  const float* Wq = (const float*)d_in[1];  const float* bq = (const float*)d_in[2];
  const float* Wk = (const float*)d_in[3];  const float* bk = (const float*)d_in[4];
  const float* Wv = (const float*)d_in[5];  const float* bv = (const float*)d_in[6];
  const float* Wo = (const float*)d_in[7];  const float* bo = (const float*)d_in[8];
  const float* W1 = (const float*)d_in[9];  const float* b1 = (const float*)d_in[10];
  const float* W2 = (const float*)d_in[11]; const float* b2 = (const float*)d_in[12];
  const float* ln1w = (const float*)d_in[13]; const float* ln1b = (const float*)d_in[14];
  const float* ln2w = (const float*)d_in[15]; const float* ln2b = (const float*)d_in[16];

  char* ws = (char*)d_ws;
  size_t off = 0;
  auto take = [&](size_t bytes) -> char* {
    char* p = ws + off;
    off = (off + bytes + 255) & ~(size_t)255;
    return p;
  };
  float*  xF   = (float*) take((size_t)NS * D * 4);
  bf16_t* xB   = (bf16_t*)take((size_t)NS * D * 2);
  float*  tmpF = (float*) take((size_t)NS * D * 4);
  bf16_t* Qb   = (bf16_t*)take((size_t)NS * D * 2);
  bf16_t* Kb   = (bf16_t*)take((size_t)NS * D * 2);
  bf16_t* VbT  = (bf16_t*)take((size_t)NS * D * 2);   // [D][NS]
  bf16_t* Yb   = (bf16_t*)take((size_t)NS * D * 2);
  bf16_t* H1   = (bf16_t*)take((size_t)NS * F * 2);
  bf16_t* WqT  = (bf16_t*)take((size_t)LAY * D * D * 2);
  bf16_t* WkT  = (bf16_t*)take((size_t)LAY * D * D * 2);
  bf16_t* WvT  = (bf16_t*)take((size_t)LAY * D * D * 2);
  bf16_t* WoT  = (bf16_t*)take((size_t)LAY * D * D * 2);
  bf16_t* W1T  = (bf16_t*)take((size_t)LAY * D * F * 2);
  bf16_t* W2T  = (bf16_t*)take((size_t)LAY * D * F * 2);

  // weights: f32 [K][N] -> bf16 [N][K]
  {
    dim3 blk(256);
    dim3 gDD2(D / 32, D / 32, LAY);
    cvt_transpose<<<gDD2, blk, 0, stream>>>(Wq, WqT, D, D);
    cvt_transpose<<<gDD2, blk, 0, stream>>>(Wk, WkT, D, D);
    cvt_transpose<<<gDD2, blk, 0, stream>>>(Wv, WvT, D, D);
    cvt_transpose<<<gDD2, blk, 0, stream>>>(Wo, WoT, D, D);
    dim3 gW1((F + 31) / 32, D / 32, LAY);
    cvt_transpose<<<gW1, blk, 0, stream>>>(W1, W1T, D, F);
    dim3 gW2(D / 32, (F + 31) / 32, LAY);
    cvt_transpose<<<gW2, blk, 0, stream>>>(W2, W2T, F, D);
  }

  {
    int total = NS * D;
    pos_encode<<<(total + 255) / 256, 256, 0, stream>>>(feat, xF, xB, total, S, D);
  }

  const dim3 gemmBlk(128);
  const dim3 gDD(D / GM_BN, NS / GM_BM);                  // N=512 (aligned)
  const dim3 gDF((F + GM_BN - 1) / GM_BN, NS / GM_BM);    // N=200 (ragged)
  const dim3 attnGrid(S / 64, H, NB);
  const int lnBlocks = (NS + 7) / 8;

  for (int l = 0; l < LAY; ++l) {
    const bf16_t* wq = WqT + (size_t)l * D * D;
    const bf16_t* wk = WkT + (size_t)l * D * D;
    const bf16_t* wv = WvT + (size_t)l * D * D;
    const bf16_t* wo = WoT + (size_t)l * D * D;
    const bf16_t* w1 = W1T + (size_t)l * D * F;
    const bf16_t* w2 = W2T + (size_t)l * D * F;

    gemm_bf16<true><<<gDD, gemmBlk, 0, stream>>>(xB, wq, bq + l * D, nullptr,
                                                 nullptr, Qb, nullptr, NS, D, D, 0);
    gemm_bf16<true><<<gDD, gemmBlk, 0, stream>>>(xB, wk, bk + l * D, nullptr,
                                                 nullptr, Kb, nullptr, NS, D, D, 0);
    gemm_bf16<true><<<gDD, gemmBlk, 0, stream>>>(xB, wv, bv + l * D, nullptr,
                                                 nullptr, nullptr, VbT, NS, D, D, 0);

    attention<<<attnGrid, gemmBlk, 0, stream>>>(Qb, Kb, VbT, Yb, S, D, NS);

    gemm_bf16<true><<<gDD, gemmBlk, 0, stream>>>(Yb, wo, bo + l * D, xF,
                                                 tmpF, nullptr, nullptr, NS, D, D, 0);
    layernorm<<<lnBlocks, 256, 0, stream>>>(tmpF, ln1w + l * D, ln1b + l * D,
                                            xF, xB, NS, D);

    gemm_bf16<false><<<gDF, gemmBlk, 0, stream>>>(xB, w1, b1 + l * F, nullptr,
                                                  nullptr, H1, nullptr, NS, F, D, 1);
    gemm_bf16<false><<<gDD, gemmBlk, 0, stream>>>(H1, w2, b2 + l * D, xF,
                                                  tmpF, nullptr, nullptr, NS, D, F, 0);
    float* outF = (l == LAY - 1) ? (float*)d_out : xF;
    layernorm<<<lnBlocks, 256, 0, stream>>>(tmpF, ln2w + l * D, ln2b + l * D,
                                            outF, xB, NS, D);
  }
}
